// SignalPropAttn5_63660005261517
// MI455X (gfx1250) — matrix operand order repaired and checked by reading, compiled, not run on hardware
//
#include <hip/hip_runtime.h>
#include <hip/hip_bf16.h>
#include <math.h>

// ---------------------------------------------------------------------------
// SignalPropAttn5 for MI455X (gfx1250).
// All dense MLP layers run on v_wmma_f32_16x16x32_f16 (f16 inputs, f32 accum).
// Weights are staged global->LDS once per workgroup, pre-swizzled into the
// WMMA B-fragment layout so each lane loads 16 contiguous halves (2x b128).
// Activations ping-pong through per-wave LDS tiles in WMMA A layout order.
// Attention is algebraically collapsed (rank-1 K/V projections).
// ---------------------------------------------------------------------------

#define NN     131072
#define ENE    262144
#define ECC    16384
#define NPIN   8192
#define NEGS   0.2f

typedef __attribute__((ext_vector_type(16))) _Float16 v16h;
typedef __attribute__((ext_vector_type(8)))  _Float16 v8h;
typedef __attribute__((ext_vector_type(8)))  float    v8f;

// Wave-level LDS ordering fence: LDS is in-order per wave on CDNA5; the
// explicit wait + scheduling barrier keeps the compiler from reordering
// cross-lane LDS produce/consume around the WMMA fragment loads.
__device__ __forceinline__ void wave_lds_fence() {
  __builtin_amdgcn_wave_barrier();
  asm volatile("s_wait_dscnt 0x0" ::: "memory");
}

// A fragment (16x32 f16, M x K) from row-major f16 LDS tile act[16][ldk].
// Lane l (m = l&15, h = l>>4): VGPR0-3 = K kt*32+h*8 .. +7, VGPR4-7 = +16..+23.
__device__ __forceinline__ v16h load_fragA(const _Float16* act, int ldk, int kt) {
  int lane = threadIdx.x & 31;
  const _Float16* p = act + (lane & 15) * ldk + kt * 32 + (lane >> 4) * 8;
  v8h lo = *(const v8h*)p;
  v8h hi = *(const v8h*)(p + 16);
  v16h a;
#pragma unroll
  for (int i = 0; i < 8; ++i) { a[i] = lo[i]; a[i + 8] = hi[i]; }
  return a;
}

// B fragment (32x16 f16, K x N) from pre-swizzled LDS: 512-half blocks per
// (kt,nt); lane's 16 halves are contiguous.
__device__ __forceinline__ v16h load_fragB(const _Float16* wsw, int ntiles, int kt, int nt) {
  int lane = threadIdx.x & 31;
  const _Float16* p = wsw + (((kt * ntiles + nt) << 9) + lane * 16);
  v8h lo = *(const v8h*)p;
  v8h hi = *(const v8h*)(p + 8);
  v16h b;
#pragma unroll
  for (int i = 0; i < 8; ++i) { b[i] = lo[i]; b[i + 8] = hi[i]; }
  return b;
}

// Stage one layer's weights W[nOut][kIn] (row-major f32, PyTorch-style) into
// swizzled f16 B-fragment order with zero padding to Kpad x Npad; bias to f32.
__device__ void stage_layer(const float* W, const float* B, int nOut, int kIn,
                            int Kpad, int Npad, _Float16* wdst, float* bdst) {
  int tid = threadIdx.x, nth = blockDim.x;
  int ntiles = Npad >> 4;
  int total = Kpad * Npad;
  for (int idx = tid; idx < total; idx += nth) {
    int k = idx / Npad, n = idx - k * Npad;
    float w = (k < kIn && n < nOut) ? W[n * kIn + k] : 0.f;   // B[k][n] = W[n][k]
    int kt = k >> 5, kin = k & 31, nt = n >> 4;
    int lane = ((kin & 8) << 1) | (n & 15);
    int hw = (kin & 7) | ((kin >> 1) & 8);
    wdst[((kt * ntiles + nt) << 9) + lane * 16 + hw] = (_Float16)w;
  }
  for (int n = tid; n < Npad; n += nth) bdst[n] = (n < nOut) ? B[n] : 0.f;
}

// One dense layer for this wave's 16-row tile: actIn[16][Kpad] (f16 LDS) x
// swizzled weights -> actOut[16][Npad] (f16 LDS), bias + optional leaky relu.
__device__ __forceinline__ void mlp_layer(const _Float16* actIn, int Kpad,
    _Float16* actOut, int Npad, const _Float16* wsw, const float* bias, bool doact)
{
  int lane = threadIdx.x & 31;
  int ktiles = Kpad >> 5, ntiles = Npad >> 4;
  int n0 = lane & 15, mb = (lane >> 4) * 8;
  for (int nt = 0; nt < ntiles; ++nt) {
    v8f acc = {};
    for (int kt = 0; kt < ktiles; ++kt) {
      v16h a = load_fragA(actIn, Kpad, kt);
      v16h b = load_fragB(wsw, ntiles, kt, nt);
      acc = __builtin_amdgcn_wmma_f32_16x16x32_f16(false, a, false, b,
                                                   (short)0, acc, false, false);
    }
    float bv = bias[nt * 16 + n0];
#pragma unroll
    for (int v = 0; v < 8; ++v) {
      float x = acc[v] + bv;
      if (doact) x = (x > 0.f) ? x : NEGS * x;
      actOut[(mb + v) * Npad + nt * 16 + n0] = (_Float16)x;
    }
  }
  wave_lds_fence();
}

// Monotone uint encoding of float for atomicMax over signed floats.
__device__ __forceinline__ unsigned encf(float x) {
  unsigned u = __float_as_uint(x);
  return (u & 0x80000000u) ? ~u : (u | 0x80000000u);
}
__device__ __forceinline__ float decf(unsigned u) {
  return (u & 0x80000000u) ? __uint_as_float(u & 0x7FFFFFFFu) : __uint_as_float(~u);
}

struct MLP5 { const float* W[5]; const float* B[5]; };
struct MLP4 { const float* W[4]; const float* B[4]; };

// ---------------------------------------------------------------------------
__global__ void __launch_bounds__(256) k_init(float* net_nf, float* nfc1,
                                              unsigned* nfc2e, int* cnt, int* pim) {
  int i = blockIdx.x * 256 + threadIdx.x;
  if (i >= NN) return;
#pragma unroll
  for (int j = 0; j < 8; ++j) net_nf[i * 8 + j] = 0.f;
#pragma unroll
  for (int j = 0; j < 32; ++j) nfc1[i * 32 + j] = 0.f;
#pragma unroll
  for (int j = 0; j < 32; ++j) nfc2e[i * 32 + j] = 0x007FFFFFu;  // encf(-inf)
  cnt[i] = 0;
  pim[i] = 0;
}

__global__ void __launch_bounds__(256) k_pi(const int* __restrict__ pi, int* pim) {
  int t = blockIdx.x * 256 + threadIdx.x;
  if (t < NPIN) pim[pi[t]] = 1;
}

// ---------------------------------------------------------------------------
// netprop: [40->128,128,128,128,->8] over EN edges; 8 waves x 16 rows / WG.
__global__ void __launch_bounds__(256) k_netprop(
    const float* __restrict__ nf, const float* __restrict__ last, MLP5 p,
    const int* __restrict__ src, const int* __restrict__ dst,
    float* __restrict__ net_nf)
{
  extern __shared__ char smem[];
  _Float16* w1 = (_Float16*)smem;                 // 64x128
  _Float16* w2 = w1 + 64 * 128;                   // 128x128
  _Float16* w3 = w2 + 128 * 128;
  _Float16* w4 = w3 + 128 * 128;
  _Float16* w5 = w4 + 128 * 128;                  // 128x16
  float* bb = (float*)(w5 + 128 * 16);
  float *b1 = bb, *b2 = bb + 128, *b3 = bb + 256, *b4 = bb + 384, *b5 = bb + 512;
  _Float16* actbase = (_Float16*)(bb + 528);      // 8 waves * 2 * 16*128
  stage_layer(p.W[0], p.B[0], 128, 40, 64, 128, w1, b1);
  stage_layer(p.W[1], p.B[1], 128, 128, 128, 128, w2, b2);
  stage_layer(p.W[2], p.B[2], 128, 128, 128, 128, w3, b3);
  stage_layer(p.W[3], p.B[3], 128, 128, 128, 128, w4, b4);
  stage_layer(p.W[4], p.B[4], 8, 128, 128, 16, w5, b5);
  __syncthreads();

  int wave = threadIdx.x >> 5, lane = threadIdx.x & 31;
  int e0 = blockIdx.x * 128 + wave * 16;
  _Float16* A  = actbase + wave * (2 * 16 * 128);
  _Float16* Bu = A + 16 * 128;

  for (int idx = lane; idx < 16 * 64; idx += 32) {   // gather x = [last|nf_s|nf_d|0pad]
    int r = idx >> 6, c = idx & 63;
    int e = e0 + r;
    float v = 0.f;
    if (c < 8)       v = last[src[e] * 8 + c];
    else if (c < 24) v = nf[src[e] * 16 + (c - 8)];
    else if (c < 40) v = nf[dst[e] * 16 + (c - 24)];
    A[r * 64 + c] = (_Float16)v;
  }
  wave_lds_fence();

  mlp_layer(A, 64, Bu, 128, w1, b1, true);
  mlp_layer(Bu, 128, A, 128, w2, b2, true);
  mlp_layer(A, 128, Bu, 128, w3, b3, true);
  mlp_layer(Bu, 128, A, 128, w4, b4, true);

  // final 128->16 (8 valid) + residual + scatter-add
  int n0 = lane & 15, mb = (lane >> 4) * 8;
  v8f acc = {};
#pragma unroll
  for (int kt = 0; kt < 4; ++kt) {
    v16h a = load_fragA(A, 128, kt);
    v16h b = load_fragB(w5, 1, kt, 0);
    acc = __builtin_amdgcn_wmma_f32_16x16x32_f16(false, a, false, b, (short)0, acc, false, false);
  }
  if (n0 < 8) {
    float bv = b5[n0];
#pragma unroll
    for (int v = 0; v < 8; ++v) {
      int e = e0 + mb + v;
      float val = acc[v] + bv + last[src[e] * 8 + n0];
      atomicAdd(&net_nf[dst[e] * 8 + n0], val);
    }
  }
}

// ---------------------------------------------------------------------------
// lutq: [40->128,128,128,->128] over EC edges -> q buffer.
__global__ void __launch_bounds__(256) k_lutq(
    const float* __restrict__ nf, const float* __restrict__ last, MLP4 p,
    const int* __restrict__ src, const int* __restrict__ dst,
    float* __restrict__ qout)
{
  extern __shared__ char smem[];
  _Float16* w1 = (_Float16*)smem;                 // 64x128
  _Float16* w2 = w1 + 64 * 128;
  _Float16* w3 = w2 + 128 * 128;
  _Float16* w4 = w3 + 128 * 128;
  float* bb = (float*)(w4 + 128 * 128);
  float *b1 = bb, *b2 = bb + 128, *b3 = bb + 256, *b4 = bb + 384;
  _Float16* actbase = (_Float16*)(bb + 512);
  stage_layer(p.W[0], p.B[0], 128, 40, 64, 128, w1, b1);
  stage_layer(p.W[1], p.B[1], 128, 128, 128, 128, w2, b2);
  stage_layer(p.W[2], p.B[2], 128, 128, 128, 128, w3, b3);
  stage_layer(p.W[3], p.B[3], 128, 128, 128, 128, w4, b4);
  __syncthreads();

  int wave = threadIdx.x >> 5, lane = threadIdx.x & 31;
  int e0 = blockIdx.x * 128 + wave * 16;
  _Float16* A  = actbase + wave * (2 * 16 * 128);
  _Float16* Bu = A + 16 * 128;

  for (int idx = lane; idx < 16 * 64; idx += 32) {
    int r = idx >> 6, c = idx & 63;
    int e = e0 + r;
    float v = 0.f;
    if (c < 8)       v = last[src[e] * 8 + c];
    else if (c < 24) v = nf[src[e] * 16 + (c - 8)];
    else if (c < 40) v = nf[dst[e] * 16 + (c - 24)];
    A[r * 64 + c] = (_Float16)v;
  }
  wave_lds_fence();

  mlp_layer(A, 64, Bu, 128, w1, b1, true);
  mlp_layer(Bu, 128, A, 128, w2, b2, true);
  mlp_layer(A, 128, Bu, 128, w3, b3, true);

  // final 128->128, no activation, straight to global (f32)
  int n0 = lane & 15, mb = (lane >> 4) * 8;
  for (int nt = 0; nt < 8; ++nt) {
    v8f acc = {};
#pragma unroll
    for (int kt = 0; kt < 4; ++kt) {
      v16h a = load_fragA(Bu, 128, kt);
      v16h b = load_fragB(w4, 8, kt, nt);
      acc = __builtin_amdgcn_wmma_f32_16x16x32_f16(false, a, false, b, (short)0, acc, false, false);
    }
    float bv = b4[nt * 16 + n0];
#pragma unroll
    for (int v = 0; v < 8; ++v)
      qout[(e0 + mb + v) * 128 + nt * 16 + n0] = acc[v] + bv;
  }
}

// ---------------------------------------------------------------------------
// attention, closed form. One thread per (edge, l).
// kh[s,:] = k[s]*Wk + bk  ==> logits = alpha*k[s] + beta (per head)
// o[h,:]  = Wv*(sum p*v) + bv (softmax sums to 1). Then @Wo^T + bo.
__global__ void __launch_bounds__(256) k_attn(
    const float* __restrict__ qbuf, const float* __restrict__ cell_ef,
    const float* Wq, const float* bq, const float* Wk, const float* bk,
    const float* Wv, const float* bv, const float* Wo, const float* bo,
    float* __restrict__ rbuf)
{
  __shared__ float sWq[1024], sWo[1024];
  __shared__ float sWk[32], sbk[32], sWv[32], sbv[32], sbq[32], sbo[32];
  for (int i = threadIdx.x; i < 1024; i += 256) { sWq[i] = Wq[i]; sWo[i] = Wo[i]; }
  if (threadIdx.x < 32) {
    int i = threadIdx.x;
    sWk[i] = Wk[i]; sbk[i] = bk[i]; sWv[i] = Wv[i]; sbv[i] = bv[i];
    sbq[i] = bq[i]; sbo[i] = bo[i];
  }
  __syncthreads();

  int t = blockIdx.x * 256 + threadIdx.x;  // t < EC*4
  int e = t >> 2, l = t & 3;

  const float* qr = qbuf + e * 128 + l * 32;
  float qh[32];
#pragma unroll
  for (int n = 0; n < 32; ++n) {
    float s = sbq[n];
#pragma unroll
    for (int k = 0; k < 32; ++k) s += qr[k] * sWq[n * 32 + k];
    qh[n] = s;
  }

  const float* axis = cell_ef + e * 256 + l * 15;  // [bias, key1*7, key2*7]
  float ka[7], kb[7];
#pragma unroll
  for (int i = 0; i < 7; ++i) { ka[i] = axis[1 + i]; kb[i] = axis[8 + i]; }
  float kp[49];
#pragma unroll
  for (int i = 0; i < 7; ++i)
#pragma unroll
    for (int j = 0; j < 7; ++j) kp[i * 7 + j] = ka[i] * kb[j];
  float kmin = kp[0], kmax = kp[0];
#pragma unroll
  for (int s = 1; s < 49; ++s) { kmin = fminf(kmin, kp[s]); kmax = fmaxf(kmax, kp[s]); }
  const float* vp = cell_ef + e * 256 + 60 + l * 49;
  float vb[49];
#pragma unroll
  for (int s = 0; s < 49; ++s) vb[s] = vp[s];

  const float scale = 0.3535533905932738f;  // 1/sqrt(8)
  float o[32];
#pragma unroll
  for (int h = 0; h < 4; ++h) {
    float alpha = 0.f, beta = 0.f;
#pragma unroll
    for (int d = 0; d < 8; ++d) {
      alpha += qh[h * 8 + d] * sWk[h * 8 + d];
      beta  += qh[h * 8 + d] * sbk[h * 8 + d];
    }
    alpha *= scale; beta *= scale;
    float mx = (alpha >= 0.f ? alpha * kmax : alpha * kmin) + beta;
    float Z = 0.f, SV = 0.f;
#pragma unroll
    for (int s = 0; s < 49; ++s) {
      float pexp = __expf(alpha * kp[s] + beta - mx);
      Z += pexp; SV += pexp * vb[s];
    }
    float sv = SV / Z;
#pragma unroll
    for (int d = 0; d < 8; ++d) o[h * 8 + d] = sWv[h * 8 + d] * sv + sbv[h * 8 + d];
  }

  float* rrow = rbuf + e * 128 + l * 32;
#pragma unroll
  for (int n = 0; n < 32; ++n) {
    float s = sbo[n];
#pragma unroll
    for (int d = 0; d < 32; ++d) s += o[d] * sWo[n * 32 + d];
    rrow[n] = s;
  }
}

// ---------------------------------------------------------------------------
// cellarc: [168->128,128,128,->73] over EC edges; gate + scatters. 4 waves/WG.
__global__ void __launch_bounds__(128) k_cellarc(
    const float* __restrict__ nf, const float* __restrict__ last,
    const float* __restrict__ rbuf, MLP4 p,
    const int* __restrict__ src, const int* __restrict__ dst,
    float* __restrict__ nfc1, unsigned* __restrict__ nfc2e,
    int* __restrict__ cnt, float* __restrict__ cef)
{
  extern __shared__ char smem[];
  _Float16* w1 = (_Float16*)smem;                 // 192x128
  _Float16* w2 = w1 + 192 * 128;
  _Float16* w3 = w2 + 128 * 128;
  _Float16* w4 = w3 + 128 * 128;                  // 128x80
  float* bb = (float*)(w4 + 128 * 80);
  float *b1 = bb, *b2 = bb + 128, *b3 = bb + 256, *b4 = bb + 384;  // b4: 80
  _Float16* actbase = (_Float16*)(bb + 464);      // 4 waves * 2 * 16*192
  float* ybase = (float*)(actbase + 4 * 2 * 16 * 192);  // 4 waves * 16*80 f32
  stage_layer(p.W[0], p.B[0], 128, 168, 192, 128, w1, b1);
  stage_layer(p.W[1], p.B[1], 128, 128, 128, 128, w2, b2);
  stage_layer(p.W[2], p.B[2], 128, 128, 128, 128, w3, b3);
  stage_layer(p.W[3], p.B[3], 73, 128, 128, 80, w4, b4);
  __syncthreads();

  int wave = threadIdx.x >> 5, lane = threadIdx.x & 31;
  int e0 = blockIdx.x * 64 + wave * 16;
  _Float16* A  = actbase + wave * (2 * 16 * 192);
  _Float16* Bu = A + 16 * 192;
  float* yb = ybase + wave * (16 * 80);

  for (int idx = lane; idx < 16 * 192; idx += 32) {  // [xe(40) | r(128) | 0pad]
    int r = idx / 192, c = idx - r * 192;
    int e = e0 + r;
    float v = 0.f;
    if (c < 8)        v = last[src[e] * 8 + c];
    else if (c < 24)  v = nf[src[e] * 16 + (c - 8)];
    else if (c < 40)  v = nf[dst[e] * 16 + (c - 24)];
    else if (c < 168) v = rbuf[e * 128 + (c - 40)];
    A[r * 192 + c] = (_Float16)v;
  }
  wave_lds_fence();

  mlp_layer(A, 192, Bu, 128, w1, b1, true);
  mlp_layer(Bu, 128, A, 128, w2, b2, true);
  mlp_layer(A, 128, Bu, 128, w3, b3, true);

  // final 128->80 (73 valid) to f32 LDS row buffer
  int n0 = lane & 15, mb = (lane >> 4) * 8;
  for (int nt = 0; nt < 5; ++nt) {
    v8f acc = {};
#pragma unroll
    for (int kt = 0; kt < 4; ++kt) {
      v16h a = load_fragA(Bu, 128, kt);
      v16h b = load_fragB(w4, 5, kt, nt);
      acc = __builtin_amdgcn_wmma_f32_16x16x32_f16(false, a, false, b, (short)0, acc, false, false);
    }
    float bv = b4[nt * 16 + n0];
#pragma unroll
    for (int v = 0; v < 8; ++v) yb[(mb + v) * 80 + nt * 16 + n0] = acc[v] + bv;
  }
  __syncthreads();

  // epilogue: 2 lanes per row; part 0 -> efc1 sum + cef, part 1 -> efc2 max + cnt
  int row = lane >> 1, part = lane & 1;
  int e = e0 + row;
  int cd = dst[e];
  const float* y = yb + (wave ? 0 : 0) + row * 80;  // yb already per-wave
  float gate = 1.f / (1.f + __expf(-y[0]));
  if (part == 0) {
#pragma unroll
    for (int j = 0; j < 32; ++j) atomicAdd(&nfc1[cd * 32 + j], y[1 + j] * gate);
#pragma unroll
    for (int j = 0; j < 8; ++j) cef[e * 8 + j] = y[65 + j];
  } else {
#pragma unroll
    for (int j = 0; j < 32; ++j) atomicMax(&nfc2e[cd * 32 + j], encf(y[33 + j] * gate));
    atomicAdd(&cnt[cd], 1);
  }
}

// ---------------------------------------------------------------------------
// cellreduce: [80->128,128,128,->8] over all N nodes + final merge.
__global__ void __launch_bounds__(256) k_cellreduce(
    const float* __restrict__ nf, const float* __restrict__ last, MLP4 p,
    const float* __restrict__ nfc1, const unsigned* __restrict__ nfc2e,
    const int* __restrict__ cnt, const int* __restrict__ pim,
    const float* __restrict__ net_nf, float* __restrict__ out)
{
  extern __shared__ char smem[];
  _Float16* w1 = (_Float16*)smem;                 // 96x128
  _Float16* w2 = w1 + 96 * 128;
  _Float16* w3 = w2 + 128 * 128;
  _Float16* w4 = w3 + 128 * 128;                  // 128x16
  float* bb = (float*)(w4 + 128 * 16);
  float *b1 = bb, *b2 = bb + 128, *b3 = bb + 256, *b4 = bb + 384;  // b4: 16
  _Float16* actbase = (_Float16*)(bb + 400);      // 8 waves * 2 * 16*128
  stage_layer(p.W[0], p.B[0], 128, 80, 96, 128, w1, b1);
  stage_layer(p.W[1], p.B[1], 128, 128, 128, 128, w2, b2);
  stage_layer(p.W[2], p.B[2], 128, 128, 128, 128, w3, b3);
  stage_layer(p.W[3], p.B[3], 8, 128, 128, 16, w4, b4);
  __syncthreads();

  int wave = threadIdx.x >> 5, lane = threadIdx.x & 31;
  int i0 = blockIdx.x * 128 + wave * 16;
  _Float16* A  = actbase + wave * (2 * 16 * 128);
  _Float16* Bu = A + 16 * 128;

  for (int idx = lane; idx < 16 * 96; idx += 32) {  // [nf | nfc1 | nfc2 | 0pad]
    int r = idx / 96, c = idx - r * 96;
    int i = i0 + r;
    float v = 0.f;
    if (c < 16)      v = nf[i * 16 + c];
    else if (c < 48) v = nfc1[i * 32 + (c - 16)];
    else if (c < 80) v = (cnt[i] > 0) ? decf(nfc2e[i * 32 + (c - 48)]) : 0.f;
    A[r * 96 + c] = (_Float16)v;
  }
  wave_lds_fence();

  mlp_layer(A, 96, Bu, 128, w1, b1, true);
  mlp_layer(Bu, 128, A, 128, w2, b2, true);
  mlp_layer(A, 128, Bu, 128, w3, b3, true);

  // final 128->16 (8 valid) + select(pi, cell vs net) -> d_out
  int n0 = lane & 15, mb = (lane >> 4) * 8;
  v8f acc = {};
#pragma unroll
  for (int kt = 0; kt < 4; ++kt) {
    v16h a = load_fragA(Bu, 128, kt);
    v16h b = load_fragB(w4, 1, kt, 0);
    acc = __builtin_amdgcn_wmma_f32_16x16x32_f16(false, a, false, b, (short)0, acc, false, false);
  }
  if (n0 < 8) {
    float bv = b4[n0];
#pragma unroll
    for (int v = 0; v < 8; ++v) {
      int i = i0 + mb + v;
      float cellv = acc[v] + bv;
      float res = pim[i] ? last[i * 8 + n0]
                         : ((cnt[i] > 0) ? cellv : net_nf[i * 8 + n0]);
      out[i * 8 + n0] = res;
    }
  }
}

// ---------------------------------------------------------------------------
extern "C" void kernel_launch(void* const* d_in, const int* in_sizes, int n_in,
                              void* d_out, int out_size, void* d_ws, size_t ws_size,
                              hipStream_t stream)
{
  (void)in_sizes; (void)n_in; (void)out_size; (void)ws_size;
  // d_in flattening: top-level dict in insertion order; MLP param lists as
  // W1,b1,...; attn dict leaves in sorted-key order (Wk,Wo,Wq,Wv,bk,bo,bq,bv).
  const float* nf      = (const float*)d_in[0];
  const float* last    = (const float*)d_in[1];   // n_atslew
  const float* cell_ef = (const float*)d_in[2];
  MLP5 np; for (int i = 0; i < 5; ++i) { np.W[i] = (const float*)d_in[3  + 2 * i]; np.B[i] = (const float*)d_in[4  + 2 * i]; }
  MLP4 lq; for (int i = 0; i < 4; ++i) { lq.W[i] = (const float*)d_in[13 + 2 * i]; lq.B[i] = (const float*)d_in[14 + 2 * i]; }
  MLP4 ca; for (int i = 0; i < 4; ++i) { ca.W[i] = (const float*)d_in[21 + 2 * i]; ca.B[i] = (const float*)d_in[22 + 2 * i]; }
  MLP4 cr; for (int i = 0; i < 4; ++i) { cr.W[i] = (const float*)d_in[29 + 2 * i]; cr.B[i] = (const float*)d_in[30 + 2 * i]; }
  const float* Wk = (const float*)d_in[37]; const float* Wo = (const float*)d_in[38];
  const float* Wq = (const float*)d_in[39]; const float* Wv = (const float*)d_in[40];
  const float* bk = (const float*)d_in[41]; const float* bo = (const float*)d_in[42];
  const float* bq = (const float*)d_in[43]; const float* bv = (const float*)d_in[44];
  const int* net_src  = (const int*)d_in[45];
  const int* net_dst  = (const int*)d_in[46];
  const int* cell_src = (const int*)d_in[47];
  const int* cell_dst = (const int*)d_in[48];
  const int* pi       = (const int*)d_in[49];

  float* ws = (float*)d_ws;
  size_t o = 0;
  float*    net_nf = ws + o; o += (size_t)NN * 8;
  float*    qbuf   = ws + o; o += (size_t)ECC * 128;
  float*    rbuf   = ws + o; o += (size_t)ECC * 128;
  float*    nfc1   = ws + o; o += (size_t)NN * 32;
  unsigned* nfc2e  = (unsigned*)(ws + o); o += (size_t)NN * 32;
  int*      cnt    = (int*)(ws + o); o += NN;
  int*      pim    = (int*)(ws + o);

  float* outnf = (float*)d_out;             // N x 8
  float* cef   = outnf + (size_t)NN * 8;    // EC x 8

  // Dynamic LDS budgets (gfx1250 WGP has 320KB):
  const int shm_np = (64*128 + 3*128*128 + 128*16) * 2 + 528 * 4 + 8 * 2 * 16 * 128 * 2;  // 186432
  const int shm_lq = (64*128 + 3*128*128) * 2 + 512 * 4 + 8 * 2 * 16 * 128 * 2;           // 182272
  const int shm_ca = (192*128 + 2*128*128 + 128*80) * 2 + 464 * 4
                   + 4 * 2 * 16 * 192 * 2 + 4 * 16 * 80 * 4;                              // 206656
  const int shm_cr = (96*128 + 2*128*128 + 128*16) * 2 + 400 * 4 + 8 * 2 * 16 * 128 * 2;  // 161344

  k_init<<<NN / 256, 256, 0, stream>>>(net_nf, nfc1, nfc2e, cnt, pim);
  k_pi<<<NPIN / 256, 256, 0, stream>>>(pi, pim);
  k_netprop<<<ENE / 128, 256, shm_np, stream>>>(nf, last, np, net_src, net_dst, net_nf);
  k_lutq<<<ECC / 128, 256, shm_lq, stream>>>(nf, last, lq, cell_src, cell_dst, qbuf);
  k_attn<<<(ECC * 4) / 256, 256, 0, stream>>>(qbuf, cell_ef, Wq, bq, Wk, bk, Wv, bv, Wo, bo, rbuf);
  k_cellarc<<<ECC / 64, 128, shm_ca, stream>>>(nf, last, rbuf, ca, cell_src, cell_dst,
                                               nfc1, nfc2e, cnt, cef);
  k_cellreduce<<<NN / 128, 256, shm_cr, stream>>>(nf, last, cr, nfc1, nfc2e, cnt, pim,
                                                  net_nf, outnf);
}
